// UnifiedLIV_43748536877630
// MI455X (gfx1250) — compile-verified, matrix operand order checked by reading, not run on hardware
//
#include <hip/hip_runtime.h>

typedef __attribute__((ext_vector_type(16))) _Float16 v16h;
typedef __attribute__((ext_vector_type(8)))  float    v8f;
typedef __attribute__((ext_vector_type(2)))  _Float16 h2;
typedef __attribute__((ext_vector_type(2)))  float    f2;
typedef __attribute__((ext_vector_type(4)))  int      i4;
typedef __attribute__((ext_vector_type(4)))  unsigned int u32x4;
typedef __attribute__((ext_vector_type(8)))  int      i32x8;
typedef __attribute__((ext_vector_type(4)))  int      i32x4;

#define BB 2
#define LL 2048
#define DD 1024
#define HH 16
#define HDD 64
#define MTOT (BB * LL)   // 4096

#define USE_ASYNC_LDS 1

#if defined(__has_builtin)
#if __has_builtin(__builtin_amdgcn_tensor_load_to_lds) && \
    __has_builtin(__builtin_amdgcn_s_wait_tensorcnt)
#define HAVE_TDM 1
#endif
#endif
#ifndef HAVE_TDM
#define HAVE_TDM 0
#endif

// K-pair pattern for 16-bit A/B WMMA fragments (CDNA5 ISA 7.12.2):
// VGPR v holds K = (v<4 ? 2v : 2(v-4)+16) + (lane>=16 ? 8 : 0), and K+1.
__device__ __forceinline__ int kpat(int v, int half8) {
    return ((v < 4) ? 2 * v : 2 * (v - 4) + 16) + half8;
}

// CDNA5 async global->LDS copy (ASYNCcnt path, CDNA5 ISA).
// LDS address = low 32 bits of the generic pointer.
__device__ __forceinline__ void async_copy_b128(const void* g, void* l) {
#if USE_ASYNC_LDS
    asm volatile("global_load_async_to_lds_b128 %0, %1, off"
                 :: "v"((unsigned int)(unsigned long long)(uintptr_t)l),
                    "v"((unsigned long long)(uintptr_t)g)
                 : "memory");
#else
    *(i4*)l = *(const i4*)g;
#endif
}
template <int N>
__device__ __forceinline__ void async_wait_le() {
#if USE_ASYNC_LDS
    asm volatile("s_wait_asynccnt %0" :: "i"(N) : "memory");
#endif
}

#if HAVE_TDM
// Tensor Data Mover 2D tile load (CDNA5 ISA/§8).
// f16 elements (data_size=1 -> 2B).  Tile rows of 64 f16 (128B) are padded in
// LDS by 16B per row (pad_interval=4 -> every 32 DWORDs, pad_amount=3 -> 4
// DWORDs), giving the 72-f16 (144B) LDS row stride the fragment readers use.
__device__ __forceinline__ void tdm_load_kv_tile(const void* gsrc, void* ldst,
                                                 int tile_rows) {
    const unsigned long long ga = (unsigned long long)(uintptr_t)gsrc;
    const unsigned int la = (unsigned int)(unsigned long long)(uintptr_t)ldst;
    u32x4 g0;
    g0[0] = 1u;                                         // count=1 (valid, user)
    g0[1] = la;                                         // lds_addr (bytes)
    g0[2] = (unsigned int)(ga & 0xFFFFFFFFu);           // global_addr[31:0]
    g0[3] = (unsigned int)((ga >> 32) & 0x01FFFFFFu)    // global_addr[56:32]
          | (2u << 30);                                 // type=2 ("image")
    i32x8 g1;
    g1[0] = (1 << 16)    // data_size = 1 (2-byte elements)
          | (1 << 20)    // pad_enable
          | (4 << 22)    // pad_interval: 32 DWORDs (= one 128B row)
          | (3 << 25);   // pad_amount:   4 DWORDs (= 16B)
    g1[1] = HDD << 16;                    // tensor_dim0 = 64 (low16); abar=0
    g1[2] = (tile_rows & 0xFFFF) << 16;   // tensor_dim0 hi=0 | tensor_dim1 lo
    g1[3] = HDD << 16;                    // tensor_dim1 hi=0 | tile_dim0 = 64
    g1[4] = tile_rows & 0xFFFF;           // tile_dim1 = rows, tile_dim2 = 0
    g1[5] = DD;                           // tensor_dim0_stride = 1024 elements
    g1[6] = 0;                            // stride hi | tensor_dim1_stride lo
    g1[7] = 0;
    const i32x4 z4 = {};
    const i32x8 z8 = {};
    __builtin_amdgcn_tensor_load_to_lds(g0, g1, z4, z4, z8, 0);
}
#endif

// ---------------------------------------------------------------------------
// Kernel 0: one-shot f32 -> f16 pack (x and the four weight matrices).
// ---------------------------------------------------------------------------
__global__ __launch_bounds__(256)
void cvt_f16_kernel(const float* __restrict__ src, _Float16* __restrict__ dst)
{
    const size_t i = ((size_t)blockIdx.x * 256 + threadIdx.x) * 4;
    f2 a = *(const f2*)(src + i);
    f2 b = *(const f2*)(src + i + 2);
    h2 ha; ha.x = (_Float16)a.x; ha.y = (_Float16)a.y;
    h2 hb; hb.x = (_Float16)b.x; hb.y = (_Float16)b.y;
    *(h2*)(dst + i)     = ha;
    *(h2*)(dst + i + 2) = hb;
}

// ---------------------------------------------------------------------------
// GEMM (f16 x f16 -> f32 accum), block tile 128x128, 256 threads = 8 waves.
// W slab (128 x 64 f16) double-buffered in LDS via async global->LDS copies.
// ---------------------------------------------------------------------------
#define KSLAB   64
#define NSLABS  (DD / KSLAB)   // 16
#define WST     72             // f16 row stride: 144B rows (16B aligned, conflict-free)
#define WCHUNKS 4              // 128 rows * 8 b128-segs / 256 threads

__device__ __forceinline__ void gemm_stage_slab(const _Float16* __restrict__ Wsrc,
                                                _Float16* __restrict__ slab,
                                                int nblock, int k, int tid)
{
#pragma unroll
    for (int u = 0; u < WCHUNKS; ++u) {
        const int c = tid + u * 256;     // 0..1023
        const int r = c >> 3;            // slab row 0..127
        const int s = c & 7;             // 16-byte segment
        async_copy_b128(Wsrc + (size_t)(nblock + r) * DD + k + s * 8,
                        slab + r * WST + s * 8);
    }
}

// ---------------------------------------------------------------------------
// Kernel 1: fused QKV projection.  Y = x @ W^T + b  (all-f16 operands)
// grid = (D/128, MTOT/128, 3)
// ---------------------------------------------------------------------------
__global__ __launch_bounds__(256)
void qkv_proj_kernel(const _Float16* __restrict__ xh,
                     const _Float16* __restrict__ WQh, const float* __restrict__ bQ,
                     const _Float16* __restrict__ WKh, const float* __restrict__ bK,
                     const _Float16* __restrict__ WVh, const float* __restrict__ bV,
                     _Float16* __restrict__ Qh, _Float16* __restrict__ Kh,
                     _Float16* __restrict__ Vh)
{
    __shared__ __attribute__((aligned(16))) _Float16 ldsW[2][128 * WST];

    const _Float16* W; const float* bias; _Float16* Y;
    if (blockIdx.z == 0)      { W = WQh; bias = bQ; Y = Qh; }
    else if (blockIdx.z == 1) { W = WKh; bias = bK; Y = Kh; }
    else                      { W = WVh; bias = bV; Y = Vh; }

    const int tid   = threadIdx.x;
    const int lane  = tid & 31;
    const int wave  = tid >> 5;
    const int nl    = lane & 15;
    const int half8 = (lane >> 4) << 3;
    const int mblock = blockIdx.y * 128;
    const int nblock = blockIdx.x * 128;
    const size_t m  = (size_t)mblock + wave * 16 + nl;

    v8f acc[8] = {};

    gemm_stage_slab(W, ldsW[0], nblock, 0, tid);          // prologue

    for (int ks = 0; ks < NSLABS; ++ks) {
        if (ks + 1 < NSLABS) {
            gemm_stage_slab(W, ldsW[(ks + 1) & 1], nblock, (ks + 1) * KSLAB, tid);
            async_wait_le<WCHUNKS>();    // oldest slab landed; next may be in flight
        } else {
            async_wait_le<0>();
        }
        __syncthreads();

        const _Float16* slab = ldsW[ks & 1];
        v16h a[2];
#pragma unroll
        for (int s = 0; s < 2; ++s)
#pragma unroll
            for (int v = 0; v < 8; ++v) {
                const int kk = kpat(v, half8) + s * 32;
                h2 av = *(const h2*)(xh + m * DD + ks * KSLAB + kk);
                a[s][2 * v] = av.x; a[s][2 * v + 1] = av.y;
            }
#pragma unroll
        for (int t = 0; t < 8; ++t) {
            const _Float16* brow = slab + (t * 16 + nl) * WST;
#pragma unroll
            for (int s = 0; s < 2; ++s) {
                v16h b;
#pragma unroll
                for (int v = 0; v < 8; ++v) {
                    const int kk = kpat(v, half8) + s * 32;
                    h2 bv = *(const h2*)(brow + kk);
                    b[2 * v] = bv.x; b[2 * v + 1] = bv.y;
                }
                acc[t] = __builtin_amdgcn_wmma_f32_16x16x32_f16(
                             false, a[s], false, b, (short)0, acc[t], false, false);
            }
        }
        __syncthreads();
    }

#pragma unroll
    for (int t = 0; t < 8; ++t) {
        const size_t n = (size_t)nblock + t * 16 + nl;
        const float bn = bias[n];
#pragma unroll
        for (int g = 0; g < 8; ++g) {
            const size_t row = (size_t)mblock + wave * 16 + g + half8;
            Y[row * DD + n] = (_Float16)(acc[t][g] + bn);
        }
    }
}

// ---------------------------------------------------------------------------
// Kernel 2: causal flash-attention.  Block = 4 waves = 64 query rows of one
// head; K/V 32x64 f16 tiles double-buffered in LDS.  Staging uses the Tensor
// Data Mover (wave 0 issues, TENSORcnt-tracked) when available, else async
// global->LDS copies (ASYNCcnt-tracked).
// grid = (L/64, B*H), block = 128
// ---------------------------------------------------------------------------
#define KVST     72   // f16 row stride: 144B (16B-aligned rows, conflict-free)
#define KVCHUNKS 4    // fallback path: (K:2 + V:2) b128 per thread per stage

__global__ __launch_bounds__(128)
void attn_kernel(const _Float16* __restrict__ Qh, const _Float16* __restrict__ Kh,
                 const _Float16* __restrict__ Vh, _Float16* __restrict__ Mh)
{
    __shared__ __attribute__((aligned(16))) _Float16 ldsK[2][32 * KVST];
    __shared__ __attribute__((aligned(16))) _Float16 ldsV[2][32 * KVST];
    __shared__ __attribute__((aligned(16))) _Float16 ldsP[4][16 * 32];

    const int lane  = threadIdx.x & 31;
    const int wave  = threadIdx.x >> 5;
    const int qtile = blockIdx.x * 4 + wave;
    const int bh    = blockIdx.y;
    const int b     = bh / HH;
    const int h     = bh % HH;
    const int q0    = qtile * 16;
    const int nl    = lane & 15;
    const int half8 = (lane >> 4) << 3;

    const _Float16* Qb = Qh + (size_t)b * LL * DD + h * HDD;
    const _Float16* Kb = Kh + (size_t)b * LL * DD + h * HDD;
    const _Float16* Vb = Vh + (size_t)b * LL * DD + h * HDD;

    auto stage_kv = [&](int buf, int jb) {
#if HAVE_TDM
        if (wave == 0) {
            tdm_load_kv_tile(Kb + (size_t)jb * DD, ldsK[buf], 32);
            tdm_load_kv_tile(Vb + (size_t)jb * DD, ldsV[buf], 32);
        }
#else
        const int c0 = threadIdx.x * 2;
#pragma unroll
        for (int u = 0; u < 2; ++u) {
            const int c = c0 + u;          // 0..255
            const int r = c >> 3;          // tile row 0..31
            const int s = c & 7;           // 16-byte segment
            async_copy_b128(Kb + (size_t)(jb + r) * DD + s * 8,
                            ldsK[buf] + r * KVST + s * 8);
            async_copy_b128(Vb + (size_t)(jb + r) * DD + s * 8,
                            ldsV[buf] + r * KVST + s * 8);
        }
#endif
    };

    // Q as two A-fragments (hd = 64 -> two k=32 steps)
    v16h aQ[2];
    {
        const size_t row = (size_t)q0 + nl;
#pragma unroll
        for (int s = 0; s < 2; ++s)
#pragma unroll
            for (int v = 0; v < 8; ++v) {
                const int kk = kpat(v, half8) + s * 32;
                h2 qv = *(const h2*)(Qb + row * DD + kk);
                aQ[s][2 * v] = qv.x; aQ[s][2 * v + 1] = qv.y;
            }
    }

    v8f o0 = {}, o1 = {}, o2 = {}, o3 = {};
    float mrow[8], lrow[8];
#pragma unroll
    for (int g = 0; g < 8; ++g) { mrow[g] = -__builtin_inff(); lrow[g] = 0.0f; }

    const float scale = 0.125f;   // 1/sqrt(64)
    const int nsteps = (blockIdx.x * 64 + 64) / 32;

    stage_kv(0, 0);               // prologue

    for (int it = 0; it < nsteps; ++it) {
        const int jb = it * 32;
        if (it + 1 < nsteps) {
            stage_kv((it + 1) & 1, jb + 32);
#if HAVE_TDM
            if (wave == 0) __builtin_amdgcn_s_wait_tensorcnt(2);
#else
            async_wait_le<KVCHUNKS>();
#endif
        } else {
#if HAVE_TDM
            if (wave == 0) __builtin_amdgcn_s_wait_tensorcnt(0);
#else
            async_wait_le<0>();
#endif
        }
        __syncthreads();

        if (jb < q0 + 16) {   // block not fully masked for this wave
            const _Float16* tK = ldsK[it & 1];
            const _Float16* tV = ldsV[it & 1];

            // ---- S = Q @ K^T for 32 key columns ----
            v8f s0 = {}, s1 = {};
#pragma unroll
            for (int s = 0; s < 2; ++s) {
                v16h bK0, bK1;
#pragma unroll
                for (int v = 0; v < 8; ++v) {
                    const int kk = kpat(v, half8) + s * 32;
                    h2 k0 = *(const h2*)(tK + nl * KVST + kk);
                    h2 k1 = *(const h2*)(tK + (16 + nl) * KVST + kk);
                    bK0[2 * v] = k0.x; bK0[2 * v + 1] = k0.y;
                    bK1[2 * v] = k1.x; bK1[2 * v + 1] = k1.y;
                }
                s0 = __builtin_amdgcn_wmma_f32_16x16x32_f16(
                         false, aQ[s], false, bK0, (short)0, s0, false, false);
                s1 = __builtin_amdgcn_wmma_f32_16x16x32_f16(
                         false, aQ[s], false, bK1, (short)0, s1, false, false);
            }

            // ---- causal mask + online softmax ----
#pragma unroll
            for (int g = 0; g < 8; ++g) {
                const int row = q0 + g + half8;
                float v0 = s0[g] * scale;
                float v1 = s1[g] * scale;
                if (jb + nl > row)      v0 = -__builtin_inff();
                if (jb + 16 + nl > row) v1 = -__builtin_inff();

                float r = fmaxf(v0, v1);
                r = fmaxf(r, __shfl_xor(r, 1, 16));
                r = fmaxf(r, __shfl_xor(r, 2, 16));
                r = fmaxf(r, __shfl_xor(r, 4, 16));
                r = fmaxf(r, __shfl_xor(r, 8, 16));

                const float mnew = fmaxf(mrow[g], r);
                const float corr = __expf(mrow[g] - mnew);
                const float p0 = __expf(v0 - mnew);
                const float p1 = __expf(v1 - mnew);

                float ps = p0 + p1;
                ps += __shfl_xor(ps, 1, 16);
                ps += __shfl_xor(ps, 2, 16);
                ps += __shfl_xor(ps, 4, 16);
                ps += __shfl_xor(ps, 8, 16);

                lrow[g] = lrow[g] * corr + ps;
                mrow[g] = mnew;
                o0[g] *= corr; o1[g] *= corr; o2[g] *= corr; o3[g] *= corr;

                ldsP[wave][(g + half8) * 32 + nl]      = (_Float16)p0;
                ldsP[wave][(g + half8) * 32 + 16 + nl] = (_Float16)p1;
            }

            // ---- reload P in A layout (same wave: DS ops in-order) ----
            v16h aP;
#pragma unroll
            for (int v = 0; v < 8; ++v) {
                const int kk = kpat(v, half8);
                h2 pv = *(const h2*)&ldsP[wave][nl * 32 + kk];
                aP[2 * v] = pv.x; aP[2 * v + 1] = pv.y;
            }

            // ---- O += P @ V (4 hd-column groups) ----
#pragma unroll
            for (int t = 0; t < 4; ++t) {
                v16h bV;
#pragma unroll
                for (int v = 0; v < 8; ++v) {
                    const int kk = kpat(v, half8);
                    bV[2 * v]     = tV[kk * KVST + t * 16 + nl];
                    bV[2 * v + 1] = tV[(kk + 1) * KVST + t * 16 + nl];
                }
                v8f& ot = (t == 0) ? o0 : (t == 1) ? o1 : (t == 2) ? o2 : o3;
                ot = __builtin_amdgcn_wmma_f32_16x16x32_f16(
                         false, aP, false, bV, (short)0, ot, false, false);
            }
        }
        __syncthreads();
    }

    // ---- normalize and write mixed (f16) ----
#pragma unroll
    for (int g = 0; g < 8; ++g) {
        const int row = q0 + g + half8;
        const float inv = 1.0f / lrow[g];
        _Float16* dst = Mh + ((size_t)b * LL + row) * DD + h * HDD;
        dst[0 * 16 + nl] = (_Float16)(o0[g] * inv);
        dst[1 * 16 + nl] = (_Float16)(o1[g] * inv);
        dst[2 * 16 + nl] = (_Float16)(o2[g] * inv);
        dst[3 * 16 + nl] = (_Float16)(o3[g] * inv);
    }
}

// ---------------------------------------------------------------------------
// Kernel 3: output projection.  y = mixed @ Wc^T  (f16 in, f32 out)
// grid = (D/128, MTOT/128)
// ---------------------------------------------------------------------------
__global__ __launch_bounds__(256)
void out_proj_kernel(const _Float16* __restrict__ Mh, const _Float16* __restrict__ Wch,
                     float* __restrict__ y)
{
    __shared__ __attribute__((aligned(16))) _Float16 ldsW[2][128 * WST];

    const int tid   = threadIdx.x;
    const int lane  = tid & 31;
    const int wave  = tid >> 5;
    const int nl    = lane & 15;
    const int half8 = (lane >> 4) << 3;
    const int mblock = blockIdx.y * 128;
    const int nblock = blockIdx.x * 128;
    const size_t m  = (size_t)mblock + wave * 16 + nl;

    v8f acc[8] = {};

    gemm_stage_slab(Wch, ldsW[0], nblock, 0, tid);

    for (int ks = 0; ks < NSLABS; ++ks) {
        if (ks + 1 < NSLABS) {
            gemm_stage_slab(Wch, ldsW[(ks + 1) & 1], nblock, (ks + 1) * KSLAB, tid);
            async_wait_le<WCHUNKS>();
        } else {
            async_wait_le<0>();
        }
        __syncthreads();

        const _Float16* slab = ldsW[ks & 1];
        v16h a[2];
#pragma unroll
        for (int s = 0; s < 2; ++s)
#pragma unroll
            for (int v = 0; v < 8; ++v) {
                const int kk = kpat(v, half8) + s * 32;
                h2 av = *(const h2*)(Mh + m * DD + ks * KSLAB + kk);
                a[s][2 * v] = av.x; a[s][2 * v + 1] = av.y;
            }
#pragma unroll
        for (int t = 0; t < 8; ++t) {
            const _Float16* brow = slab + (t * 16 + nl) * WST;
#pragma unroll
            for (int s = 0; s < 2; ++s) {
                v16h b;
#pragma unroll
                for (int v = 0; v < 8; ++v) {
                    const int kk = kpat(v, half8) + s * 32;
                    h2 bv = *(const h2*)(brow + kk);
                    b[2 * v] = bv.x; b[2 * v + 1] = bv.y;
                }
                acc[t] = __builtin_amdgcn_wmma_f32_16x16x32_f16(
                             false, a[s], false, b, (short)0, acc[t], false, false);
            }
        }
        __syncthreads();
    }

#pragma unroll
    for (int t = 0; t < 8; ++t) {
        const size_t n = (size_t)nblock + t * 16 + nl;
#pragma unroll
        for (int g = 0; g < 8; ++g) {
            const size_t row = (size_t)mblock + wave * 16 + g + half8;
            y[row * DD + n] = acc[t][g];
        }
    }
}

// ---------------------------------------------------------------------------
extern "C" void kernel_launch(void* const* d_in, const int* in_sizes, int n_in,
                              void* d_out, int out_size, void* d_ws, size_t ws_size,
                              hipStream_t stream) {
    const float* x  = (const float*)d_in[0];
    const float* WQ = (const float*)d_in[1];
    const float* bQ = (const float*)d_in[2];
    const float* WK = (const float*)d_in[3];
    const float* bK = (const float*)d_in[4];
    const float* WV = (const float*)d_in[5];
    const float* bV = (const float*)d_in[6];
    const float* Wc = (const float*)d_in[7];
    float* y = (float*)d_out;

    // workspace layout (f16):
    //   xh (4M) | WQh | WKh | WVh | Wch (1M each) | Qh | Kh | Vh | Mh (4M each)
    _Float16* xh  = (_Float16*)d_ws;
    _Float16* WQh = xh  + (size_t)MTOT * DD;
    _Float16* WKh = WQh + (size_t)DD * DD;
    _Float16* WVh = WKh + (size_t)DD * DD;
    _Float16* Wch = WVh + (size_t)DD * DD;
    _Float16* Qh  = Wch + (size_t)DD * DD;
    _Float16* Kh  = Qh  + (size_t)MTOT * DD;
    _Float16* Vh  = Kh  + (size_t)MTOT * DD;
    _Float16* Mh  = Vh  + (size_t)MTOT * DD;

    cvt_f16_kernel<<<(MTOT * DD) / 1024, 256, 0, stream>>>(x,  xh);
    cvt_f16_kernel<<<(DD * DD) / 1024,   256, 0, stream>>>(WQ, WQh);
    cvt_f16_kernel<<<(DD * DD) / 1024,   256, 0, stream>>>(WK, WKh);
    cvt_f16_kernel<<<(DD * DD) / 1024,   256, 0, stream>>>(WV, WVh);
    cvt_f16_kernel<<<(DD * DD) / 1024,   256, 0, stream>>>(Wc, Wch);

    dim3 gproj(DD / 128, MTOT / 128, 3);
    qkv_proj_kernel<<<gproj, 256, 0, stream>>>(xh, WQh, bQ, WKh, bK, WVh, bV,
                                               Qh, Kh, Vh);

    dim3 gattn(LL / 64, BB * HH);
    attn_kernel<<<gattn, 128, 0, stream>>>(Qh, Kh, Vh, Mh);

    dim3 gout(DD / 128, MTOT / 128);
    out_proj_kernel<<<gout, 256, 0, stream>>>(Mh, Wch, y);
}
